// GRUModel_19224273616952
// MI455X (gfx1250) — matrix operand order repaired and checked by reading, not compiled, not run on hardware
//
#include <hip/hip_runtime.h>
#include <hip/hip_bf16.h>
#include <cstdint>
#include <cstddef>

// ---------------- problem constants ----------------
static constexpr int T_  = 200;
static constexpr int B_  = 128;
static constexpr int N_  = 50;
static constexpr int H_  = 512;
static constexpr int P_  = 10;
static constexpr int DIN = 3 * N_;          // 150
static constexpr int DINP = 160;            // padded to mult of 32
static constexpr int G3  = 3 * H_;          // 1536
static constexpr int OUT_ = 2 * N_;         // 100
static constexpr int OUTP = 112;            // padded to mult of 16
static constexpr int KHP  = 544;            // (H+1)=513 padded to mult of 32
static constexpr float DECAY_CAP = 1000.0f;

// ---------------- WMMA types ----------------
typedef __bf16 v16bf __attribute__((ext_vector_type(16)));
typedef float  v8f   __attribute__((ext_vector_type(8)));

union FragBF { v16bf v; uint4 q[2]; };

// A-fragment (16x32 bf16, row-major source with row stride ldk elements).
// ISA layout: lane l -> M=l&15 ; lane<16 holds K = k0+{0..7,16..23},
// lane>=16 holds K = k0+{8..15,24..31}.  -> two 16B loads.
__device__ inline v16bf load_a_frag(const __bf16* A, int ldk, int k0, int lane) {
    FragBF f;
    const __bf16* p = A + (lane & 15) * ldk + k0 + ((lane >> 4) << 3);
    f.q[0] = *reinterpret_cast<const uint4*>(p);
    f.q[1] = *reinterpret_cast<const uint4*>(p + 16);
    return f.v;
}

// B-fragment (32x16 bf16, row-major K x N, row stride ldn elements).
// ISA layout: lane l -> N col n0+(l&15); lane<16 holds K = k0+{0..7,16..23},
// lane>=16 holds K = k0+{8..15,24..31}.
__device__ inline v16bf load_b_frag(const __bf16* Bm, int ldn, int k0, int n0, int lane) {
    FragBF f;
    const __bf16* p = Bm + (size_t)(k0 + ((lane >> 4) << 3)) * ldn + n0 + (lane & 15);
    for (int s = 0; s < 16; ++s)
        f.v[s] = p[(size_t)((s & 7) + ((s >> 3) << 4)) * ldn];
    return f.v;
}

__device__ inline v8f wmma_bf16(v16bf a, v16bf b, v8f c) {
    return __builtin_amdgcn_wmma_f32_16x16x32_bf16(false, a, false, b, (short)0, c, false, false);
}

__device__ inline float decay_f(float dt, float w) {
    float a = fminf(fmaxf(dt * w, 0.0f), DECAY_CAP);
    return expf(-a);
}
__device__ inline float sigm_f(float x) { return 1.0f / (1.0f + expf(-x)); }

// ---------------- small prep kernels ----------------

// all_dt (T,B) and pred_dt (T,B,P) from t (B,T)
__global__ void build_dt_kernel(const float* __restrict__ tm,
                                float* __restrict__ allDt,
                                float* __restrict__ predDt) {
    int i = blockIdx.x * blockDim.x + threadIdx.x;
    if (i >= T_ * B_) return;
    int ti = i / B_, b = i - ti * B_;
    float tcur = tm[b * T_ + ti];
    allDt[i] = (ti == 0) ? tcur : (tcur - tm[b * T_ + ti - 1]);
    for (int p = 0; p < P_; ++p) {
        int idx = ti + 1 + p;
        float valid = (idx < T_) ? 1.0f : 0.0f;
        int idxc = idx < (T_ - 1) ? idx : (T_ - 1);
        predDt[(size_t)i * P_ + p] = (tm[b * T_ + idxc] - tcur) * valid;
    }
}

// X1bf (T*B, 160) = bf16 concat(y, delta_t_node, mask), zero-padded K
__global__ void build_x1_kernel(const float* __restrict__ y,
                                const float* __restrict__ mask,
                                const float* __restrict__ dtn,
                                __bf16* __restrict__ X1) {
    size_t i = (size_t)blockIdx.x * blockDim.x + threadIdx.x;
    size_t total = (size_t)T_ * B_ * DINP;
    if (i >= total) return;
    size_t row = i / DINP; int k = (int)(i - row * DINP);
    float v = 0.0f;
    if (k < N_)            v = y[row * N_ + k];
    else if (k < 2 * N_)   v = dtn[row * N_ + (k - N_)];
    else if (k < 3 * N_)   v = mask[row * N_ + (k - 2 * N_)];
    X1[i] = (__bf16)v;
}

// dst (Kpad x Npad) bf16 = transpose of src (Nsrc x Ksrc) f32, zero padded
__global__ void transpose_bf_kernel(const float* __restrict__ src,
                                    __bf16* __restrict__ dst,
                                    int Nsrc, int Ksrc, int Npad, int Kpad) {
    size_t i = (size_t)blockIdx.x * blockDim.x + threadIdx.x;
    size_t total = (size_t)Kpad * Npad;
    if (i >= total) return;
    int k = (int)(i / Npad), n = (int)(i - (size_t)k * Npad);
    float v = (k < Ksrc && n < Nsrc) ? src[(size_t)n * Ksrc + k] : 0.0f;
    dst[i] = (__bf16)v;
}

// ---------------- generic bf16 GEMM: C(MxN f32) = A(MxK bf16) @ B(KxN bf16) + bias ----------------
// block: 256 threads (8 waves). grid: (M/16, N/128). K mult of 32, K<=512.
// K-loop software-pipelined: next fragments are loaded before the current WMMA.
__global__ void gemm_bf16_kernel(const __bf16* __restrict__ A,
                                 const __bf16* __restrict__ Bm,
                                 const float* __restrict__ bias,
                                 float* __restrict__ C,
                                 int M, int N, int K) {
    __shared__ __align__(16) __bf16 As[16 * 520];
    const int ldk = K + 8;
    const int m0 = blockIdx.x * 16;
    for (int e = threadIdx.x; e < 16 * K; e += 256) {
        int r = e / K, k = e - r * K;
        As[r * ldk + k] = A[(size_t)(m0 + r) * K + k];
    }
    __syncthreads();
    const int wave = threadIdx.x >> 5, lane = threadIdx.x & 31;
    const int n0 = blockIdx.y * 128 + wave * 16;
    if (n0 >= N) return;
    v8f acc = {0.f, 0.f, 0.f, 0.f, 0.f, 0.f, 0.f, 0.f};
    v16bf a0 = load_a_frag(As, ldk, 0, lane);
    v16bf b0 = load_b_frag(Bm, N, 0, n0, lane);
    for (int k0 = 32; k0 < K; k0 += 32) {
        v16bf a1 = load_a_frag(As, ldk, k0, lane);
        v16bf b1 = load_b_frag(Bm, N, k0, n0, lane);
        acc = wmma_bf16(a0, b0, acc);
        a0 = a1; b0 = b1;
    }
    acc = wmma_bf16(a0, b0, acc);
    const int hi = lane >> 4, jn = n0 + (lane & 15);
    const float bv = bias ? bias[jn] : 0.0f;
    for (int v = 0; v < 8; ++v) {
        int m = m0 + v + hi * 8;
        C[(size_t)m * N + jn] = acc[v] + bv;
    }
}

// ---------------- fused GRU-D step ----------------
// One launch per (layer, t). Computes h_new (B x H) f32 + bf16 history row.
// block 256 (8 waves), grid (8, 4): M-band of 16 batch rows x quarter of H.
// K-loop software-pipelined across the 4 fragments (A, Br, Bz, Bh).
__global__ void grud_step_kernel(const float* __restrict__ hprev, int first,
                                 const float* __restrict__ Wx, int t,
                                 const __bf16* __restrict__ Ut,   // (H x 3H) bf16
                                 const float* __restrict__ dw,
                                 const float* __restrict__ allDt,
                                 float* __restrict__ hout,        // (B x H) f32
                                 __bf16* __restrict__ houtbf) {   // (T*B x H) bf16 history
    __shared__ __align__(16) __bf16 As[16 * 520];
    const int ldk = 520;
    const int m0 = blockIdx.x * 16;
    // stage decayed h_prev band (16 x 512) as bf16
    for (int e = threadIdx.x; e < 16 * H_; e += 256) {
        int r = e >> 9, k = e & (H_ - 1);
        int b = m0 + r;
        float dt = allDt[t * B_ + b];
        float v = first ? 0.0f : hprev[(size_t)b * H_ + k];
        As[r * ldk + k] = (__bf16)(v * decay_f(dt, dw[k]));
    }
    __syncthreads();
    const int wave = threadIdx.x >> 5, lane = threadIdx.x & 31;
    const int n0 = blockIdx.y * 128 + wave * 16;   // j tile within H
    v8f ar = {0.f,0.f,0.f,0.f,0.f,0.f,0.f,0.f};
    v8f az = ar, ah = ar;
    v16bf a0  = load_a_frag(As, ldk, 0, lane);
    v16bf br0 = load_b_frag(Ut, G3, 0, n0,          lane);
    v16bf bz0 = load_b_frag(Ut, G3, 0, H_ + n0,     lane);
    v16bf bh0 = load_b_frag(Ut, G3, 0, 2 * H_ + n0, lane);
    for (int k0 = 32; k0 < H_; k0 += 32) {
        v16bf a1  = load_a_frag(As, ldk, k0, lane);
        v16bf br1 = load_b_frag(Ut, G3, k0, n0,          lane);
        v16bf bz1 = load_b_frag(Ut, G3, k0, H_ + n0,     lane);
        v16bf bh1 = load_b_frag(Ut, G3, k0, 2 * H_ + n0, lane);
        ar = wmma_bf16(a0, br0, ar);
        az = wmma_bf16(a0, bz0, az);
        ah = wmma_bf16(a0, bh0, ah);
        a0 = a1; br0 = br1; bz0 = bz1; bh0 = bh1;
    }
    ar = wmma_bf16(a0, br0, ar);
    az = wmma_bf16(a0, bz0, az);
    ah = wmma_bf16(a0, bh0, ah);
    const int hi = lane >> 4, jn = n0 + (lane & 15);
    const float dwj = dw[jn];
    for (int v = 0; v < 8; ++v) {
        int b = m0 + v + hi * 8;
        size_t row = (size_t)t * B_ + b;
        float wr = Wx[row * G3 + jn];
        float wz = Wx[row * G3 + H_ + jn];
        float wh = Wx[row * G3 + 2 * H_ + jn];
        float dt = allDt[t * B_ + b];
        float hd = first ? 0.0f : hprev[(size_t)b * H_ + jn] * decay_f(dt, dwj);
        float r = sigm_f(wr + ar[v]);
        float z = sigm_f(wz + az[v]);
        float ht = tanhf(wh + ah[v] * r);
        float hn = hd + z * (ht - hd);
        hout[(size_t)b * H_ + jn] = hn;
        houtbf[row * H_ + jn] = (__bf16)hn;
    }
}

// ---------------- prediction head ----------------
// rows = T*B*P ; A row = [h2*decay(pred_dt), pred_dt, 0pad] (K=544), B = fc_Wt (544x112)
// block 256 (8 waves, 7 compute N-tiles), grid (T*B*P/16). Pipelined K-loop.
__global__ void head_kernel(const float* __restrict__ h2,
                            const float* __restrict__ predDt,
                            const float* __restrict__ dw2,
                            const __bf16* __restrict__ fcWt,
                            const float* __restrict__ fcb,
                            float* __restrict__ outp) {
    __shared__ __align__(16) __bf16 As[16 * 552];
    const int ldk = 552;
    const int m0 = blockIdx.x * 16;
    for (int e = threadIdx.x; e < 16 * KHP; e += 256) {
        int r = e / KHP, k = e - r * KHP;
        int row = m0 + r;
        int t = row / (B_ * P_);
        int rem = row - t * (B_ * P_);
        int b = rem / P_, p = rem - b * P_;
        float pdt = predDt[((size_t)t * B_ + b) * P_ + p];
        float v;
        if (k < H_)        v = h2[((size_t)t * B_ + b) * H_ + k] * decay_f(pdt, dw2[k]);
        else if (k == H_)  v = pdt;
        else               v = 0.0f;
        As[r * ldk + k] = (__bf16)v;
    }
    __syncthreads();
    const int wave = threadIdx.x >> 5, lane = threadIdx.x & 31;
    if (wave >= 7) return;                 // 7 x 16 = 112 padded output cols
    const int n0 = wave * 16;
    v8f acc = {0.f,0.f,0.f,0.f,0.f,0.f,0.f,0.f};
    v16bf a0 = load_a_frag(As, ldk, 0, lane);
    v16bf b0 = load_b_frag(fcWt, OUTP, 0, n0, lane);
    for (int k0 = 32; k0 < KHP; k0 += 32) {
        v16bf a1 = load_a_frag(As, ldk, k0, lane);
        v16bf b1 = load_b_frag(fcWt, OUTP, k0, n0, lane);
        acc = wmma_bf16(a0, b0, acc);
        a0 = a1; b0 = b1;
    }
    acc = wmma_bf16(a0, b0, acc);
    const int hi = lane >> 4, jn = n0 + (lane & 15);
    if (jn >= OUT_) return;
    const float bv = fcb[jn];
    for (int v = 0; v < 8; ++v) {
        int row = m0 + v + hi * 8;
        outp[(size_t)row * OUT_ + jn] = acc[v] + bv;
    }
}

// ---------------- workspace layout (bytes, all 256-aligned) ----------------
static constexpr size_t OFF_ALLDT = 0;                              // T*B f32          = 102400
static constexpr size_t OFF_PREDDT = OFF_ALLDT + 102400;            // T*B*P f32        = 1024000
static constexpr size_t OFF_X1BF  = OFF_PREDDT + 1024000;           // 25600*160 bf16   = 8192000
static constexpr size_t OFF_W1T   = OFF_X1BF + 8192000;             // 160*1536 bf16    = 491520
static constexpr size_t OFF_U1T   = OFF_W1T + 491520;               // 512*1536 bf16    = 1572864
static constexpr size_t OFF_W2T   = OFF_U1T + 1572864;
static constexpr size_t OFF_U2T   = OFF_W2T + 1572864;
static constexpr size_t OFF_FCWT  = OFF_U2T + 1572864;              // 544*112 bf16     = 121856
static constexpr size_t OFF_WX    = OFF_FCWT + 121856;              // 25600*1536 f32   = 157286400
static constexpr size_t OFF_H1BF  = OFF_WX + 157286400;             // 25600*512 bf16   = 26214400
static constexpr size_t OFF_HP0   = OFF_H1BF + 26214400;            // 128*512 f32      = 262144
static constexpr size_t OFF_HP1   = OFF_HP0 + 262144;
static constexpr size_t OFF_H2F   = OFF_HP1 + 262144;               // 25600*512 f32    = 52428800
// total = 251,104,256 bytes

extern "C" void kernel_launch(void* const* d_in, const int* in_sizes, int n_in,
                              void* d_out, int out_size, void* d_ws, size_t ws_size,
                              hipStream_t stream) {
    const float* tm   = (const float*)d_in[0];   // (B,T)
    const float* y    = (const float*)d_in[1];   // (T,B,N)
    const float* mask = (const float*)d_in[2];   // (T,B,N)
    const float* dtn  = (const float*)d_in[3];   // (T,B,N)
    // d_in[4] = init_state (zeros) -> dtg == 0 identically, unused
    const float* W1   = (const float*)d_in[5];   // (3H, DIN)
    const float* b1   = (const float*)d_in[6];   // (3H,)
    const float* U1   = (const float*)d_in[7];   // (3H, H)
    const float* W2   = (const float*)d_in[8];
    const float* b2   = (const float*)d_in[9];
    const float* U2   = (const float*)d_in[10];
    const float* dw1  = (const float*)d_in[11];  // (H,)
    const float* dw2  = (const float*)d_in[12];  // (H,)
    const float* fcW  = (const float*)d_in[13];  // (OUT, H+1)
    const float* fcb  = (const float*)d_in[14];  // (OUT,)
    float* outp = (float*)d_out;

    char* ws = (char*)d_ws;
    float*  allDt  = (float*)(ws + OFF_ALLDT);
    float*  predDt = (float*)(ws + OFF_PREDDT);
    __bf16* X1bf   = (__bf16*)(ws + OFF_X1BF);
    __bf16* W1t    = (__bf16*)(ws + OFF_W1T);
    __bf16* U1t    = (__bf16*)(ws + OFF_U1T);
    __bf16* W2t    = (__bf16*)(ws + OFF_W2T);
    __bf16* U2t    = (__bf16*)(ws + OFF_U2T);
    __bf16* fcWt   = (__bf16*)(ws + OFF_FCWT);
    float*  Wx     = (float*)(ws + OFF_WX);      // reused for layer 1 then layer 2
    __bf16* h1bf   = (__bf16*)(ws + OFF_H1BF);   // layer-1 history (bf16); scratch for layer 2
    float*  hp[2]  = { (float*)(ws + OFF_HP0), (float*)(ws + OFF_HP1) };
    float*  h2f    = (float*)(ws + OFF_H2F);     // layer-2 full f32 history

    const int M = T_ * B_;               // 25600
    const int MH = M * P_;               // 256000 head rows

    // --- prep ---
    build_dt_kernel<<<(T_ * B_ + 255) / 256, 256, 0, stream>>>(tm, allDt, predDt);
    {
        size_t tot = (size_t)M * DINP;
        build_x1_kernel<<<(unsigned)((tot + 255) / 256), 256, 0, stream>>>(y, mask, dtn, X1bf);
    }
    transpose_bf_kernel<<<(DINP * G3 + 255) / 256, 256, 0, stream>>>(W1, W1t, G3, DIN, G3, DINP);
    transpose_bf_kernel<<<(H_ * G3 + 255) / 256, 256, 0, stream>>>(U1, U1t, G3, H_, G3, H_);
    transpose_bf_kernel<<<(H_ * G3 + 255) / 256, 256, 0, stream>>>(W2, W2t, G3, H_, G3, H_);
    transpose_bf_kernel<<<(H_ * G3 + 255) / 256, 256, 0, stream>>>(U2, U2t, G3, H_, G3, H_);
    transpose_bf_kernel<<<(KHP * OUTP + 255) / 256, 256, 0, stream>>>(fcW, fcWt, OUT_, H_ + 1, OUTP, KHP);

    // --- Wx1 = X1 @ W1^T + b1 ---
    gemm_bf16_kernel<<<dim3(M / 16, G3 / 128), 256, 0, stream>>>(X1bf, W1t, b1, Wx, M, G3, DINP);

    // --- layer 1 scan ---
    for (int t = 0; t < T_; ++t) {
        const float* hprev = hp[(t ^ 1) & 1];
        float* hout = hp[t & 1];
        grud_step_kernel<<<dim3(B_ / 16, H_ / 128), 256, 0, stream>>>(
            hprev, (t == 0) ? 1 : 0, Wx, t, U1t, dw1, allDt, hout, h1bf);
    }

    // --- Wx2 = h1 @ W2^T + b2 ---
    gemm_bf16_kernel<<<dim3(M / 16, G3 / 128), 256, 0, stream>>>(h1bf, W2t, b2, Wx, M, G3, H_);

    // --- layer 2 scan (full f32 history; bf16 sink reuses h1bf, now dead) ---
    for (int t = 0; t < T_; ++t) {
        const float* hprev = (t == 0) ? h2f : (h2f + (size_t)(t - 1) * B_ * H_);
        float* hout = h2f + (size_t)t * B_ * H_;
        grud_step_kernel<<<dim3(B_ / 16, H_ / 128), 256, 0, stream>>>(
            hprev, (t == 0) ? 1 : 0, Wx, t, U2t, dw2, allDt, hout, h1bf);
    }

    // --- prediction head ---
    head_kernel<<<MH / 16, 256, 0, stream>>>(h2f, predDt, dw2, fcWt, fcb, outp);
}
